// ExpertsChooseMaskedMLP_25348896981199
// MI455X (gfx1250) — compile-verified, matrix operand order checked by reading, not run on hardware
//
#include <hip/hip_runtime.h>
#include <hip/hip_bf16.h>
#include <math.h>

// ---------------------------------------------------------------------------
// ExpertsChooseMaskedMLP on gfx1250 (MI455X):
//   Phase 0: one-shot f32->bf16 convert (+ transpose where the einsum needs a
//            k-strided operand) so every GEMM operand is k-contiguous bf16.
//   Phase 1: 4 batched GEMMs with V_WMMA_F32_16X16X32_BF16.
//            - LDS staged via GLOBAL_LOAD_ASYNC_TO_LDS_B128 (ASYNCcnt)
//            - double-buffered LDS: tile i+1 DMA overlaps tile i WMMAs
//            - 128x64 block tile, 8 waves x 4 accumulators -> 8 WMMA/wave/K64
//            - all tile fragments hoisted to registers, then WMMA burst
//            - GELU/bias fused in epilogues; one barrier per K-tile
// ---------------------------------------------------------------------------

typedef __attribute__((ext_vector_type(16))) __bf16 v16bf;
typedef __attribute__((ext_vector_type(8)))  __bf16 v8bf;
typedef __attribute__((ext_vector_type(8)))  float  v8f;

__device__ __forceinline__ __bf16 toBf(float f) {
    union { float f; unsigned u; } v; v.f = f;
    unsigned r = v.u + 0x7fffu + ((v.u >> 16) & 1u);   // round-to-nearest-even
    unsigned short h = (unsigned short)(r >> 16);
    __bf16 b;
    __builtin_memcpy(&b, &h, 2);
    return b;
}
__device__ __forceinline__ __bf16 toBf(__bf16 x) { return x; }

__device__ __forceinline__ float geluExact(float x) {
    return 0.5f * x * (1.0f + erff(x * 0.70710678118654752440f));
}

// Low 32 bits of a generic pointer into the LDS aperture == LDS byte offset
// (ISA 10.2: the upper address bits are dropped).
__device__ __forceinline__ unsigned ldsOff(const void* p) {
    return (unsigned)(unsigned long long)p;
}

// One per-lane 16-byte async copy: LDS[vdst] = MEM[vaddr] (ASYNCcnt-tracked).
__device__ __forceinline__ void asyncCopyB128(unsigned lds, const void* g) {
    asm volatile("global_load_async_to_lds_b128 %0, %1, off"
                 :: "v"(lds), "v"((unsigned long long)g)
                 : "memory");
}
__device__ __forceinline__ void waitAsync0() {
    asm volatile("s_wait_asynccnt 0x0" ::: "memory");
}

constexpr int BM   = 128;  // block tile rows (8 waves x 16)
constexpr int BN   = 64;   // block tile cols (4 subtiles of 16 per wave)
constexpr int BK   = 64;   // K per LDS stage (2 WMMA k-steps of 32)
constexpr int LDSK = 72;   // padded LDS row (144B, multiple of 16B)

// ---------------------------------------------------------------------------
// Phase-0 kernels
// ---------------------------------------------------------------------------

__global__ __launch_bounds__(256)
void convert_kernel(const float* __restrict__ in, __bf16* __restrict__ out, long n) {
    long i = ((long)blockIdx.x * 256 + threadIdx.x) * 8;
    if (i + 8 > n) return;
    v8bf o;
    #pragma unroll
    for (int j = 0; j < 8; ++j) o[j] = toBf(in[i + j]);
    *(v8bf*)(out + i) = o;
}

// out[plane][col][row] = toBf(in[plane][row][col]); input col-contiguous.
template <typename TI>
__global__ __launch_bounds__(256)
void transpose_conv_kernel(const TI* __restrict__ in, long iOut, long iIn, long sR,
                           __bf16* __restrict__ out, long oPlane, long oLd, int mod)
{
    __shared__ __bf16 tile[32][34];
    const int p = blockIdx.z;
    const long q = p / mod, rm = p % mod;
    const TI* ip = in + q * iOut + rm * iIn;
    __bf16* op = out + (long)p * oPlane;

    const int tx = threadIdx.x, ty = threadIdx.y;     // block (32,8)
    const int c0 = blockIdx.x * 32, r0 = blockIdx.y * 32;

    #pragma unroll
    for (int i = 0; i < 4; ++i)
        tile[ty + 8 * i][tx] = toBf(ip[(long)(r0 + ty + 8 * i) * sR + (c0 + tx)]);
    __syncthreads();
    #pragma unroll
    for (int i = 0; i < 4; ++i)
        op[(long)(c0 + ty + 8 * i) * oLd + (r0 + tx)] = tile[tx][ty + 8 * i];
}

// ---------------------------------------------------------------------------
// WMMA GEMM: C[m][n] = sum_k A[m][k]*B[n][k], A/B bf16 k-contiguous.
// EPI: 0 = store bf16, 1 = +bias[n]+GELU bf16, 2 = +bias[n] bf16, 3 = f32
// ---------------------------------------------------------------------------

__device__ __forceinline__ v16bf frag16(const __bf16* lo, const __bf16* hi) {
    v8bf a = *(const v8bf*)lo;
    v8bf b = *(const v8bf*)hi;
    v16bf r;
    #pragma unroll
    for (int j = 0; j < 8; ++j) { r[j] = a[j]; r[8 + j] = b[j]; }
    return r;
}

template <int EPI>
__global__ __launch_bounds__(256)
void gemm_wmma_kernel(const __bf16* __restrict__ Aall, long ldA, long aOut, long aIn,
                      const __bf16* __restrict__ Ball, long ldB, long bOut, long bIn,
                      void* __restrict__ outAll, long ldOut, long oOut, long oIn,
                      const float* __restrict__ biasAll, long biasIn,
                      int K, int bMod)
{
    __shared__ alignas(16) __bf16 Alds[2][BM][LDSK];   // A[m][k], double buffered
    __shared__ alignas(16) __bf16 Blds[2][BN][LDSK];   // B[n][k], double buffered

    const int bz = blockIdx.z;
    const long q = bz / bMod, rm = bz % bMod;

    const __bf16* A  = Aall + q * aOut + rm * aIn;
    const __bf16* Bp = Ball + q * bOut + rm * bIn;
    const float* bias = biasAll ? (biasAll + rm * biasIn) : nullptr;
    const long outBase = q * oOut + rm * oIn;

    const int tid  = threadIdx.x;
    const int lane = tid & 31;
    const int wave = tid >> 5;     // 0..7: wave owns rows [wave*16, wave*16+16)
    const int m0   = blockIdx.y * BM;
    const int n0   = blockIdx.x * BN;

    // staging coords: A 128x64 -> 32 elems/thread; B 64x64 -> 16 elems/thread
    const int ar = tid >> 1, ak = (tid & 1) * 32;   // 4x b128
    const int br = tid >> 2, bk = (tid & 3) * 16;   // 2x b128

    const __bf16* gaRow = A  + (long)(m0 + ar) * ldA + ak;
    const __bf16* gbRow = Bp + (long)(n0 + br) * ldB + bk;
    const unsigned aL[2] = { ldsOff(&Alds[0][ar][ak]), ldsOff(&Alds[1][ar][ak]) };
    const unsigned bL[2] = { ldsOff(&Blds[0][br][bk]), ldsOff(&Blds[1][br][bk]) };

    auto stageTile = [&](int buf, int k0) {
        const __bf16* ga = gaRow + k0;
        const __bf16* gb = gbRow + k0;
        asyncCopyB128(aL[buf],      ga);
        asyncCopyB128(aL[buf] + 16, ga + 8);
        asyncCopyB128(aL[buf] + 32, ga + 16);
        asyncCopyB128(aL[buf] + 48, ga + 24);
        asyncCopyB128(bL[buf],      gb);
        asyncCopyB128(bL[buf] + 16, gb + 8);
        if (k0 + BK < K) {                       // prefetch one tile further
            __builtin_prefetch(ga + BK, 0, 1);
            __builtin_prefetch(gb + BK, 0, 1);
        }
    };

    v8f acc[4] = {{}, {}, {}, {}};

    const int half = lane >> 4;         // K-subgroup select (ISA A/B layout)
    const int r    = lane & 15;

    // prologue: fill buffer 0
    stageTile(0, 0);
    waitAsync0();
    __syncthreads();

    int cur = 0;
    for (int k0 = 0; k0 < K; k0 += BK) {
        if (k0 + BK < K) stageTile(cur ^ 1, k0 + BK);   // DMA overlaps WMMAs

        // Hoist ALL fragments for this K-tile into registers first so the
        // compiler batches the 20 ds_load_b128 + one dscnt wait, then issues
        // the 8 WMMAs back-to-back (no per-WMMA ds_load serialization).
        v16bf af[2];
        v16bf bf[2][4];
        #pragma unroll
        for (int kk = 0; kk < 2; ++kk) {
            const int kb = kk * 32;
            af[kk] = frag16(&Alds[cur][wave * 16 + r][kb + half * 8],
                            &Alds[cur][wave * 16 + r][kb + 16 + half * 8]);
            #pragma unroll
            for (int nt = 0; nt < 4; ++nt)
                bf[kk][nt] = frag16(&Blds[cur][nt * 16 + r][kb + half * 8],
                                    &Blds[cur][nt * 16 + r][kb + 16 + half * 8]);
        }
        #pragma unroll
        for (int kk = 0; kk < 2; ++kk)
            #pragma unroll
            for (int nt = 0; nt < 4; ++nt)
                acc[nt] = __builtin_amdgcn_wmma_f32_16x16x32_bf16(
                    false, af[kk], false, bf[kk][nt], (short)0, acc[nt], false, false);

        waitAsync0();        // own next-buffer copies complete
        __syncthreads();     // all waves: copies landed + done reading `cur`
        cur ^= 1;
    }

    // D layout: lane l, VGPR v -> M = v + 8*(l>=16), N = l&15
    const int rowOff = 8 * (lane >> 4);
    #pragma unroll
    for (int nt = 0; nt < 4; ++nt) {
        const int gcol = n0 + nt * 16 + r;
        #pragma unroll
        for (int v = 0; v < 8; ++v) {
            const int grow = m0 + wave * 16 + rowOff + v;
            float val = acc[nt][v];
            if (EPI == 1) val = geluExact(val + bias[gcol]);
            if (EPI == 2) val = val + bias[gcol];
            const long idx = outBase + (long)grow * ldOut + gcol;
            if (EPI == 3) ((float*)outAll)[idx] = val;
            else          ((__bf16*)outAll)[idx] = toBf(val);
        }
    }
}

// ---------------------------------------------------------------------------

extern "C" void kernel_launch(void* const* d_in, const int* in_sizes, int n_in,
                              void* d_out, int out_size, void* d_ws, size_t ws_size,
                              hipStream_t stream)
{
    (void)in_sizes; (void)n_in; (void)out_size; (void)ws_size;

    const float* x  = (const float*)d_in[0];   // [B,T,D]
    const float* dm = (const float*)d_in[1];   // [B,T,E,C]
    const float* cb = (const float*)d_in[2];   // [B,T,E,C]
    const float* W1 = (const float*)d_in[3];   // [E,O1,D]
    const float* b1 = (const float*)d_in[4];   // [E,O1]
    const float* W2 = (const float*)d_in[5];   // [E,D,O1]
    const float* b2 = (const float*)d_in[6];   // [D]
    float* out = (float*)d_out;                // [B,T,D]

    constexpr int B = 2, T = 2048, D = 2048, E = 8, C = 512, O1 = D / E;

    // workspace layout (bf16), with dead-buffer reuse (~138 MB total)
    const size_t nDmT = (size_t)B * E * C * T;   // dmT  [B,E,C,T]   (later: y)
    const size_t nXT  = (size_t)B * D * T;       // xT   [B,D,T]
    const size_t nCb  = (size_t)B * T * E * C;   // cbB  [B,T,E*C]
    const size_t nW1  = (size_t)E * O1 * D;      // W1b  [E,O1,D]
    const size_t nW2  = (size_t)E * D * O1;      // W2b  [E,D,O1]
    const size_t nXe  = (size_t)B * E * C * D;   // xe   [B,E,C,D]   (later: yT)

    __bf16* dmT = (__bf16*)d_ws;
    __bf16* xT  = dmT + nDmT;
    __bf16* cbB = xT  + nXT;
    __bf16* W1b = cbB + nCb;
    __bf16* W2b = W1b + nW1;
    __bf16* xe  = W2b + nW2;
    __bf16* h   = xe  + nXe;                     // [B,E,C,O1]
    __bf16* y   = dmT;                           // reuse (dmT dead after G1)
    __bf16* yT  = xe;                            // reuse (xe  dead after G2)

    dim3 blk(256), tblk(32, 8);

    // --- Phase 0: convert / transpose-convert ---
    convert_kernel<<<(int)(nCb / 2048), blk, 0, stream>>>(cb, cbB, (long)nCb);
    convert_kernel<<<(int)(nW1 / 2048), blk, 0, stream>>>(W1, W1b, (long)nW1);
    convert_kernel<<<(int)(nW2 / 2048), blk, 0, stream>>>(W2, W2b, (long)nW2);
    // dmT[b,e][c][t] = dm[b,t,e,c]
    transpose_conv_kernel<float><<<dim3(C / 32, T / 32, B * E), tblk, 0, stream>>>(
        dm, (long)T * E * C, (long)C, (long)E * C, dmT, (long)C * T, (long)T, E);
    // xT[b][d][t] = x[b,t,d]
    transpose_conv_kernel<float><<<dim3(D / 32, T / 32, B), tblk, 0, stream>>>(
        x, (long)T * D, 0L, (long)D, xT, (long)D * T, (long)T, 1);

    // --- G1: xe[b,e][c,d] = sum_t dmT[c,t] * xT[d,t]   (M=C,N=D,K=T) ---
    gemm_wmma_kernel<0><<<dim3(D / BN, C / BM, B * E), blk, 0, stream>>>(
        dmT, (long)T, (long)E * C * T, (long)C * T,
        xT,  (long)T, (long)D * T,     0L,
        xe,  (long)D, (long)E * C * D, (long)C * D,
        nullptr, 0L, /*K*/T, /*bMod*/E);

    // --- G2: h = gelu(xe @ W1^T + b1)   (M=C,N=O1,K=D) ---
    gemm_wmma_kernel<1><<<dim3(O1 / BN, C / BM, B * E), blk, 0, stream>>>(
        xe,  (long)D, (long)E * C * D, (long)C * D,
        W1b, (long)D, 0L,              (long)O1 * D,
        h,   (long)O1, (long)E * C * O1, (long)C * O1,
        b1, (long)O1, /*K*/D, /*bMod*/E);

    // --- G3: y = h @ W2^T + b2   (M=C,N=D,K=O1) ---
    gemm_wmma_kernel<2><<<dim3(D / BN, C / BM, B * E), blk, 0, stream>>>(
        h,   (long)O1, (long)E * C * O1, (long)C * O1,
        W2b, (long)O1, 0L,               (long)D * O1,
        y,   (long)D, (long)E * C * D,   (long)C * D,
        b2, 0L, /*K*/O1, /*bMod*/E);

    // yT[b][o][e*C+c] = y[b,e,c,o]
    transpose_conv_kernel<__bf16><<<dim3(D / 32, (E * C) / 32, B), tblk, 0, stream>>>(
        y, (long)E * C * D, 0L, (long)D, yT, (long)D * E * C, (long)E * C, 1);

    // --- G4: out[b][t,o] = sum_ec cbB[t,ec] * yT[o,ec]   (M=T,N=D,K=E*C) ---
    gemm_wmma_kernel<3><<<dim3(D / BN, T / BM, B), blk, 0, stream>>>(
        cbB, (long)E * C, (long)T * E * C, 0L,
        yT,  (long)E * C, (long)D * E * C, 0L,
        out, (long)D, (long)T * D, 0L,
        nullptr, 0L, /*K*/E * C, /*bMod*/1);
}
